// DTWLayer_85014582657726
// MI455X (gfx1250) — matrix operand order repaired and checked by reading, not compiled, not run on hardware
//
#include <hip/hip_runtime.h>
#include <hip/hip_fp16.h>

// Problem constants (from the reference)
#define BB   16     // batch
#define DD   16     // feature dim
#define TT   1024   // time steps
#define PP   64     // patterns
#define LL   64     // pattern length
#define TILE 64     // time-tile width
#define NTILE (TT / TILE)
#define BIGV    1e30f
#define HALFBIG 5e29f
#define WDEC 0.96466155f   // 0.1^(1/64)

typedef __attribute__((ext_vector_type(16))) _Float16 v16h;
typedef __attribute__((ext_vector_type(8)))  float    v8f;

// One workgroup (one wave32) per (b,p) pair.
//  - WMMA produces xp tiles; the epilogue converts them to f16 DTW distances
//    in LDS (keeps sqrt OFF the serial DP chain, halves LDS footprint).
//  - The same wave runs the DTW DP as an anti-diagonal wavefront:
//    lane λ owns rows i0=2λ, i1=2λ+1; at step s it processes column j=s-λ.
//    D[i-1,j] comes from lane λ-1 via shfl_up of its previous-step value.
__launch_bounds__(32)
__global__ void dtw_fused_kernel(const float* __restrict__ x,
                                 const float* __restrict__ patts,
                                 float* __restrict__ out) {
  const int p    = blockIdx.x;
  const int b    = blockIdx.y;
  const int lane = threadIdx.x;

  __shared__ float  pT[LL][DD];              // patts[p] as [l][d]
  __shared__ float  x2b[TILE];               // ||x||^2 per t of current tile
  __shared__ __half distb[2][TILE][LL + 4];  // f16 distances [slot][t][l]

  // ---- load patts[p] transposed: patts layout (p, d, l) ----
  for (int i = lane; i < LL * DD; i += 32) {
    int l = i >> 4, d = i & 15;
    pT[l][d] = patts[(p * DD + d) * LL + l];
  }
  __syncthreads();

  const int  mloc   = lane & 15;
  const bool hiHalf = (lane >= 16);
  const int  kbase  = hiHalf ? 8 : 0;  // A layout: lanes 16-31 hold K=8..15

  // ---- B fragments (32x16 f16, K padded 16->32) + per-lane column norms ----
  // lanes 0-15: e=0..7 hold B[k=e][n]; lanes 16-31: e=0..7 hold B[k=8+e][n]; e=8..15 (K=16..31) = 0
  v16h  bf[4];
  float p2col[4];
#pragma unroll
  for (int tn = 0; tn < 4; ++tn) {
    int n = tn * 16 + mloc;
    float s = 0.f;
#pragma unroll
    for (int e = 0; e < 16; ++e) {
      float v = pT[n][e];
      s += v * v;
      bf[tn][e] = (e < 8) ? (_Float16)pT[n][kbase + e] : (_Float16)0.f;
    }
    p2col[tn] = s;   // ||patts[p,:,n]||^2, used in the WMMA epilogue
  }

  // ---- DP state (column j-1 values for this lane's two rows) ----
  float d0 = BIGV;      // D[i0, j-1]
  float d1 = BIGV;      // D[i1, j-1]
  float hprev = BIGV;   // D[i0-1, j-1] (from lane above, one step stale)
  const int i0 = 2 * lane, i1 = i0 + 1;
  const int outBase = ((b * PP + p) * LL) * TILE;  // out[b][p][l][tau]

  // advance both rows of this lane at column j, given h = D[i0-1, j]
  auto cell = [&](int j, float h, bool doStore) {
    int sl = (j >> 6) & 1;
    int tl = j & (TILE - 1);
    __half2 dv = *(const __half2*)&distb[sl][tl][i0];  // 4B-aligned (i0 even)
    float dist0 = __low2float(dv);
    float dist1 = __high2float(dv);
    float m0 = fminf(fminf(h, d0), hprev);
    m0 = (m0 >= HALFBIG) ? 0.f : m0;
    float n0 = fmaf(WDEC, m0, dist0);
    float m1 = fminf(fminf(n0, d1), d0);   // up=n0, left=d1, diag=old d0
    m1 = (m1 >= HALFBIG) ? 0.f : m1;
    float n1 = fmaf(WDEC, m1, dist1);
    d0 = n0;
    d1 = n1;
    if (doStore && j >= TT - TILE) {       // last 64 columns -> output
      int tau = j - (TT - TILE);
      out[outBase + i0 * TILE + tau] = n0;
      out[outBase + i1 * TILE + tau] = n1;
    }
  };

  for (int tile = 0; tile < NTILE; ++tile) {
    const int t0   = tile * TILE;
    const int slot = tile & 1;

    // prefetch next x tile (16 rows x 256B, one 128B half-row per lane)
    if (tile + 1 < NTILE) {
      const float* pf =
          x + (b * DD + (lane & 15)) * TT + (t0 + TILE) + ((lane >> 4) * 32);
      __builtin_prefetch(pf, 0, 1);
    }

    // ---- ||x||^2 for this tile (coalesced along t across lanes) ----
    for (int t = lane; t < TILE; t += 32) {
      float s = 0.f;
#pragma unroll
      for (int d = 0; d < DD; ++d) {
        float v = x[(b * DD + d) * TT + t0 + t];
        s += v * v;
      }
      x2b[t] = s;
    }
    __syncthreads();

    // ---- WMMA: xp tile, epilogue converts straight to f16 distances ----
#pragma unroll
    for (int tm = 0; tm < 4; ++tm) {
      // A fragment: rows m = tm*16 + (lane&15), K=16..31 zero-padded;
      // loads coalesce across lanes (consecutive t within a d-row)
      const float* xrow = x + (b * DD + kbase) * TT + t0 + (tm * 16 + mloc);
      v16h af;
#pragma unroll
      for (int e = 0; e < 8; ++e) af[e] = (_Float16)xrow[e * TT];
#pragma unroll
      for (int e = 8; e < 16; ++e) af[e] = (_Float16)0.f;
      float x2row[8];
#pragma unroll
      for (int r = 0; r < 8; ++r)
        x2row[r] = x2b[tm * 16 + (hiHalf ? r + 8 : r)];
#pragma unroll
      for (int tn = 0; tn < 4; ++tn) {
        v8f acc = {};
        acc = __builtin_amdgcn_wmma_f32_16x16x32_f16(
            /*neg_a=*/false, af, /*neg_b=*/false, bf[tn],
            /*c_mod=*/(short)0, acc, /*reuse_a=*/false, /*reuse_b=*/false);
        // D layout: VGPR r -> row r (lanes<16) / r+8 (lanes>=16), col = lane&15
#pragma unroll
        for (int r = 0; r < 8; ++r) {
          float sq = fmaf(-2.f, acc[r], x2row[r] + p2col[tn]);
          // raw v_sqrt_f32: ~2 ULP f32 error, invisible after f16 rounding;
          // avoids the IEEE sqrt expansion (range scale + Newton tail)
          float dv = __builtin_amdgcn_sqrtf(fmaxf(sq, 1e-12f));
          int   mr = hiHalf ? r + 8 : r;
          distb[slot][tm * 16 + mr][tn * 16 + mloc] = __float2half(dv);
        }
      }
    }
    __syncthreads();

    // ---- consume: 64 wavefront steps; columns span tiles (tile-1, tile) ----
    const bool lastTile = (tile == NTILE - 1);
    if (tile == 0) {
      for (int ss = 0; ss < TILE; ++ss) {
        int   j = ss - lane;
        float h = __shfl_up(d1, 1, 32);
        if (lane == 0) h = BIGV;
        if (j >= 0) cell(j, h, false);
        hprev = h;
      }
    } else {
      // j = t0+ss-lane >= 64-31 > 0: no bounds check on the hot path
#pragma unroll 4
      for (int ss = 0; ss < TILE; ++ss) {
        int   j = t0 + ss - lane;
        float h = __shfl_up(d1, 1, 32);
        if (lane == 0) h = BIGV;
        cell(j, h, lastTile);
        hprev = h;
      }
    }
  }

  // ---- drain: lanes 1..31 finish their last columns (j in [993,1023]) ----
  for (int s = TT; s < TT + 31; ++s) {
    int   j = s - lane;
    float h = __shfl_up(d1, 1, 32);
    if (lane == 0) h = BIGV;
    if (j < TT) cell(j, h, true);
    hprev = h;
  }
}

extern "C" void kernel_launch(void* const* d_in, const int* in_sizes, int n_in,
                              void* d_out, int out_size, void* d_ws, size_t ws_size,
                              hipStream_t stream) {
  (void)in_sizes; (void)n_in; (void)out_size; (void)d_ws; (void)ws_size;
  const float* x     = (const float*)d_in[0];   // (16, 16, 1024)
  const float* patts = (const float*)d_in[1];   // (64, 16, 64)
  float*       out   = (float*)d_out;           // (16, 64, 64, 64)
  dim3 grid(PP, BB);   // one workgroup (1 wave) per (pattern, batch) pair
  dim3 block(32);
  hipLaunchKernelGGL(dtw_fused_kernel, grid, block, 0, stream, x, patts, out);
}
